// GCN_87351044866139
// MI455X (gfx1250) — matrix operand order repaired and checked by reading, not compiled, not run on hardware
//
#include <hip/hip_runtime.h>
#include <hip/hip_bf16.h>
#include <math.h>

#define D 128   // D_IN == HID == 128

typedef __attribute__((ext_vector_type(16))) _Float16 v16h;
typedef __attribute__((ext_vector_type(8)))  _Float16 v8h;
typedef __attribute__((ext_vector_type(8)))  float    v8f;

// ---------------------------------------------------------------------------
// Elementwise helpers
// ---------------------------------------------------------------------------
__global__ void fill_f32(float* p, float v, long n) {
    long i = (long)blockIdx.x * blockDim.x + threadIdx.x;
    if (i < n) p[i] = v;
}

__global__ void accum_deg(const int* __restrict__ dst, float* __restrict__ deg, int E) {
    int e = blockIdx.x * blockDim.x + threadIdx.x;
    if (e < E) atomicAdd(&deg[dst[e]], 1.0f);
}

__global__ void make_dinv(float* __restrict__ deg, int n) {
    int i = blockIdx.x * blockDim.x + threadIdx.x;
    if (i < n) {
        float d = deg[i];
        deg[i] = (d > 0.0f) ? rsqrtf(d) : 0.0f;   // deg >= 1 here (self loops)
    }
}

// Vectorized f32 -> f16 convert (8 elements / thread: 2x b128 load, 1x b128 store)
__global__ void f32_to_f16(const float* __restrict__ in, _Float16* __restrict__ out, long n) {
    long i = ((long)blockIdx.x * blockDim.x + threadIdx.x) * 8;
    if (i >= n) return;
    const float4 q0 = *(const float4*)(in + i);
    const float4 q1 = *(const float4*)(in + i + 4);
    v8h o;
    o[0] = (_Float16)q0.x; o[1] = (_Float16)q0.y; o[2] = (_Float16)q0.z; o[3] = (_Float16)q0.w;
    o[4] = (_Float16)q1.x; o[5] = (_Float16)q1.y; o[6] = (_Float16)q1.z; o[7] = (_Float16)q1.w;
    *(v8h*)(out + i) = o;
}

// ---------------------------------------------------------------------------
// WMMA GEMM: Y[n_rows x 128](f32) = X16[n_rows x 128](f16) @ W[128 x 128](f32)
//
// * A operand arrives pre-converted f16: each fragment is two contiguous
//   16-byte global_load_b128s per lane -> no conversions in the hot loop.
// * W staged in LDS pre-swizzled into the WMMA B register layout
//   (ISA 7.12.2: lanes 0-15 hold K 0-7,16-23, lanes 16-31 hold K 8-15,24-31):
//   each B fragment is one contiguous 32-byte read (2x ds_load_b128).
// * Each wave owns TWO 16-row M-tiles: every B fragment feeds two
//   v_wmma_f32_16x16x32_f16 => 64 WMMAs/wave.
// ---------------------------------------------------------------------------
__global__ __launch_bounds__(256) void gcn_gemm_wmma(const _Float16* __restrict__ X,
                                                     const float* __restrict__ W,
                                                     float* __restrict__ Y,
                                                     int n_rows) {
    // Wl flat index: ((nt*4 + c)*32 + lane)*16 + i
    //   maps to W[k][n] with k = c*32 + (i>>3)*16 + (lane>>4)*8 + (i&7), n = nt*16 + (lane&15)
    __shared__ __align__(32) _Float16 Wl[D * D];

    const int tid = threadIdx.x;
    for (int f = tid; f < D * D; f += 256) {
        int i  = f & 15;
        int ln = (f >> 4) & 31;
        int c  = (f >> 9) & 3;
        int nt = f >> 11;
        int k  = c * 32 + ((i >> 3) << 4) + ((ln >> 4) << 3) + (i & 7);
        int n  = nt * 16 + (ln & 15);
        Wl[f] = (_Float16)W[k * D + n];
    }
    __syncthreads();

    const int wave = tid >> 5;
    const int lane = tid & 31;
    const int m0 = (blockIdx.x * 8 + wave) * 32;   // 32 rows per wave
    if (m0 >= n_rows) return;

    // ---- A operands: lane l holds row (l&15); lanes 0-15 take K
    //      kbase..kbase+7 / +16..+23, lanes 16-31 are +8. ----
    const int kbase = (lane >> 4) * 8;

    v16h a[2][4];
#pragma unroll
    for (int t = 0; t < 2; ++t) {
        int row = m0 + t * 16 + (lane & 15);
        if (row >= n_rows) row = n_rows - 1;       // clamped (store is guarded)
        const _Float16* xr = X + (size_t)row * D;
#pragma unroll
        for (int c = 0; c < 4; ++c) {
            const v8h lo = *(const v8h*)(xr + c * 32 + kbase);
            const v8h hi = *(const v8h*)(xr + c * 32 + 16 + kbase);
            a[t][c] = __builtin_shufflevector(lo, hi,
                         0, 1, 2, 3, 4, 5, 6, 7, 8, 9, 10, 11, 12, 13, 14, 15);
        }
    }

    const int col    = lane & 15;
    const bool full0 = (m0 + 16) <= n_rows;
    const bool full1 = (m0 + 32) <= n_rows;
    const bool any1  = (m0 + 16) <  n_rows;

#pragma unroll
    for (int nt = 0; nt < 8; ++nt) {
        v8f acc0 = {}, acc1 = {};
#pragma unroll
        for (int c = 0; c < 4; ++c) {
            const v16h b = *(const v16h*)&Wl[((nt * 4 + c) * 32 + lane) * 16];
            acc0 = __builtin_amdgcn_wmma_f32_16x16x32_f16(
                false, a[0][c], false, b, (short)0, acc0, false, false);
            acc1 = __builtin_amdgcn_wmma_f32_16x16x32_f16(
                false, a[1][c], false, b, (short)0, acc1, false, false);
        }
        // ---- D: f32 16x16, VGPR r -> M = r + (lane>>4)*8, N = lane&15 ----
        float* yc = Y + (size_t)(m0 + ((lane >> 4) * 8)) * D + nt * 16 + col;
        if (full0) {
#pragma unroll
            for (int r = 0; r < 8; ++r) yc[(size_t)r * D] = acc0[r];
        } else {
#pragma unroll
            for (int r = 0; r < 8; ++r) {
                int m = m0 + r + ((lane >> 4) * 8);
                if (m < n_rows) yc[(size_t)r * D] = acc0[r];
            }
        }
        if (full1) {
#pragma unroll
            for (int r = 0; r < 8; ++r) yc[(size_t)(16 + r) * D] = acc1[r];
        } else if (any1) {
#pragma unroll
            for (int r = 0; r < 8; ++r) {
                int m = m0 + 16 + r + ((lane >> 4) * 8);
                if (m < n_rows) yc[(size_t)(16 + r) * D] = acc1[r];
            }
        }
    }
}

// ---------------------------------------------------------------------------
// Edge scatter-add: agg[dst] += h[src] * dinv[src]*dinv[dst]
// One wave per edge; float4 gather per lane, 4x global_atomic_add_f32.
// ---------------------------------------------------------------------------
__global__ __launch_bounds__(256) void edge_aggregate(const float* __restrict__ h,
                                                      const int* __restrict__ src,
                                                      const int* __restrict__ dst,
                                                      const float* __restrict__ dinv,
                                                      float* __restrict__ agg,
                                                      int E) {
    long tid  = (long)blockIdx.x * blockDim.x + threadIdx.x;
    long e    = tid >> 5;
    int  lane = (int)(tid & 31);
    if (e >= E) return;
    int s = src[e], d = dst[e];
    float nrm = dinv[s] * dinv[d];
    const float4 v = *(const float4*)(h + (size_t)s * D + lane * 4);
    float* ap = agg + (size_t)d * D + lane * 4;
    atomicAdd(ap + 0, v.x * nrm);
    atomicAdd(ap + 1, v.y * nrm);
    atomicAdd(ap + 2, v.z * nrm);
    atomicAdd(ap + 3, v.w * nrm);
}

// ---------------------------------------------------------------------------
// Finalize: agg[i][k] = relu(agg[i][k] + hlin[i][k]*dinv[i]^2 + bias[k])
// Optionally writes an f16 copy (next layer's WMMA A operand) fused in.
// ---------------------------------------------------------------------------
__global__ void gcn_finalize(const float* __restrict__ hlin,
                             float* __restrict__ agg,
                             const float* __restrict__ dinv,
                             const float* __restrict__ bias,
                             _Float16* __restrict__ out16,   // may be null
                             int n) {
    long i = (long)blockIdx.x * blockDim.x + threadIdx.x;
    long total = (long)n * D;
    if (i >= total) return;
    long node = i >> 7;            // /128
    int  k    = (int)(i & (D - 1));
    float di = dinv[node];
    float v = agg[i] + hlin[i] * di * di + bias[k];
    v = v > 0.0f ? v : 0.0f;
    agg[i] = v;
    if (out16) out16[i] = (_Float16)v;
}

// ---------------------------------------------------------------------------
// Edge classifier: wave per edge, shuffle reduction, sigmoid.
// ---------------------------------------------------------------------------
__global__ __launch_bounds__(256) void edge_classify(const float* __restrict__ h,
                                                     const int* __restrict__ src,
                                                     const int* __restrict__ dst,
                                                     const float* __restrict__ Wc,
                                                     const float* __restrict__ bc,
                                                     float* __restrict__ probs,
                                                     int E) {
    long tid  = (long)blockIdx.x * blockDim.x + threadIdx.x;
    long e    = tid >> 5;
    int  lane = (int)(tid & 31);
    if (e >= E) return;
    int s = src[e], d = dst[e];

    float acc0 = 0.0f, acc1 = 0.0f;
    const float4 hs4 = *(const float4*)(h + (size_t)s * D + lane * 4);
    const float4 hd4 = *(const float4*)(h + (size_t)d * D + lane * 4);
    const float hs[4] = {hs4.x, hs4.y, hs4.z, hs4.w};
    const float hd[4] = {hd4.x, hd4.y, hd4.z, hd4.w};
#pragma unroll
    for (int j = 0; j < 4; ++j) {
        int k = lane * 4 + j;
        acc0 += hs[j] * Wc[k * 2 + 0] + hd[j] * Wc[(D + k) * 2 + 0];
        acc1 += hs[j] * Wc[k * 2 + 1] + hd[j] * Wc[(D + k) * 2 + 1];
    }
#pragma unroll
    for (int off = 16; off > 0; off >>= 1) {
        acc0 += __shfl_down(acc0, off, 32);
        acc1 += __shfl_down(acc1, off, 32);
    }
    if (lane == 0) {
        probs[e * 2 + 0] = 1.0f / (1.0f + expf(-(acc0 + bc[0])));
        probs[e * 2 + 1] = 1.0f / (1.0f + expf(-(acc1 + bc[1])));
    }
}

// ---------------------------------------------------------------------------
// Host launch
// ---------------------------------------------------------------------------
extern "C" void kernel_launch(void* const* d_in, const int* in_sizes, int n_in,
                              void* d_out, int out_size, void* d_ws, size_t ws_size,
                              hipStream_t stream) {
    const float* feats = (const float*)d_in[0];
    const int*   eidx  = (const int*)d_in[1];   // [2, E] flat
    const float* W1    = (const float*)d_in[2];
    const float* b1    = (const float*)d_in[3];
    const float* W2    = (const float*)d_in[4];
    const float* b2    = (const float*)d_in[5];
    const float* Wc    = (const float*)d_in[6];
    const float* bc    = (const float*)d_in[7];

    const int N = in_sizes[0] / D;      // 50000
    const int E = in_sizes[1] / 2;      // 800000
    const int* src = eidx;
    const int* dst = eidx + E;

    float* out_h     = (float*)d_out;                    // [N, 128]
    float* out_probs = out_h + (size_t)N * D;            // [E, 2]

    // workspace layout
    float*    dinv = (float*)d_ws;                       // N f32
    float*    bufA = dinv + N;                           // N*128 f32 (GEMM out)
    float*    bufB = bufA + (size_t)N * D;               // N*128 f32 (agg)
    _Float16* x16  = (_Float16*)(bufB + (size_t)N * D);  // N*128 f16 (GEMM in)

    const long ND = (long)N * D;
    const int  TB = 256;
    dim3 blk(TB);
    auto cdiv = [](long a, long b) { return (unsigned)((a + b - 1) / b); };

    const unsigned gemm_grid = cdiv(cdiv(N, 32), 8);     // 8 waves/block, 32 rows/wave
    const unsigned edge_grid = cdiv((long)E * 32, TB);   // wave per edge

    // --- degrees / symmetric norm ---
    fill_f32<<<cdiv(N, TB), blk, 0, stream>>>(dinv, 1.0f, N);       // self loop
    accum_deg<<<cdiv(E, TB), blk, 0, stream>>>(dst, dinv, E);
    make_dinv<<<cdiv(N, TB), blk, 0, stream>>>(dinv, N);

    // --- layer 1: bufB = relu(scatter(feats@W1) + selfloop + b1); x16 = f16(bufB) ---
    f32_to_f16<<<cdiv(ND / 8, TB), blk, 0, stream>>>(feats, x16, ND);
    gcn_gemm_wmma<<<gemm_grid, blk, 0, stream>>>(x16, W1, bufA, N);
    fill_f32<<<cdiv(ND, TB), blk, 0, stream>>>(bufB, 0.0f, ND);
    edge_aggregate<<<edge_grid, blk, 0, stream>>>(bufA, src, dst, dinv, bufB, E);
    gcn_finalize<<<cdiv(ND, TB), blk, 0, stream>>>(bufA, bufB, dinv, b1, x16, N);

    // --- layer 2: out_h = relu(scatter(h1@W2) + selfloop + b2) ---
    gcn_gemm_wmma<<<gemm_grid, blk, 0, stream>>>(x16, W2, bufA, N);
    fill_f32<<<cdiv(ND, TB), blk, 0, stream>>>(out_h, 0.0f, ND);
    edge_aggregate<<<edge_grid, blk, 0, stream>>>(bufA, src, dst, dinv, out_h, E);
    gcn_finalize<<<cdiv(ND, TB), blk, 0, stream>>>(bufA, out_h, dinv, b2, (_Float16*)nullptr, N);

    // --- edge classifier ---
    edge_classify<<<edge_grid, blk, 0, stream>>>(out_h, src, dst, Wc, bc, out_probs, E);
}
